// EfficientMixedScoreMultiHeadAttentionLayer_54528904790830
// MI455X (gfx1250) — matrix operand order repaired and checked by reading, not compiled, not run on hardware
//
#include <hip/hip_runtime.h>
#include <hip/hip_bf16.h>
#include <math.h>

// Problem constants (match reference)
#define BB 4
#define RR 256
#define CC 256
#define EE 512
#define HH 16
#define DD 32
#define NORM_F 0.17677669529663687f   // 1/sqrt(32)
#define TANH_CLIP_F 10.0f

typedef __attribute__((ext_vector_type(16))) __bf16 bf16x16;
typedef __attribute__((ext_vector_type(8)))  float  f32x8;
typedef __attribute__((ext_vector_type(4)))  int    i32x4;

#define F32X8_ZERO {0.f,0.f,0.f,0.f,0.f,0.f,0.f,0.f}

// gfx1250 async global->LDS path (ASYNCcnt); compile-safe detection
#if defined(__gfx1250__) && __has_builtin(__builtin_amdgcn_global_load_async_to_lds_b128) && __has_builtin(__builtin_amdgcn_s_wait_asynccnt)
#define USE_ASYNC_LDS 1
typedef __attribute__((address_space(1))) i32x4* as1_i32x4_p;
typedef __attribute__((address_space(3))) i32x4* as3_i32x4_p;
#else
#define USE_ASYNC_LDS 0
#endif

static __device__ __forceinline__ __bf16 f2bf(float f) {
  unsigned u = __float_as_uint(f);
  unsigned r = u + 0x7FFFu + ((u >> 16) & 1u);   // round-to-nearest-even
  unsigned short h = (unsigned short)(r >> 16);
  __bf16 b;
  __builtin_memcpy(&b, &h, 2);
  return b;
}

static __device__ __forceinline__ f32x8 wmma_bf16(bf16x16 a, bf16x16 b, f32x8 c) {
  // 8 args: (neg_a, A, neg_b, B, c_mod, C, reuse_a, reuse_b)
  return __builtin_amdgcn_wmma_f32_16x16x32_bf16(false, a, false, b, (short)0, c,
                                                 false, false);
}

union Frag32B { uint4 q[2]; bf16x16 v; };

// A fragment 16x32 (MxK), row-major source. Contiguous k order used here: lanes 0-15: M=lane,
// K=0..15; lanes 16-31: M=lane-16, K=16..31.
static __device__ __forceinline__ bf16x16 load_a_frag_g(const __bf16* A, int lda, int lane) {
  const __bf16* p = A + (size_t)(lane & 15) * lda + ((lane >> 4) << 4);
  Frag32B u;
  u.q[0] = *(const uint4*)(p);        // elements k.. k+7
  u.q[1] = *(const uint4*)(p + 8);    // elements k+8..k+15
  return u.v;
}

// Same layout, element-wise (for LDS)
static __device__ __forceinline__ bf16x16 load_a_frag_e(const __bf16* A, int lda, int lane) {
  const __bf16* p = A + (size_t)(lane & 15) * lda + ((lane >> 4) << 4);
  bf16x16 r;
#pragma unroll
  for (int i = 0; i < 8; ++i) { r[i] = p[i]; r[8 + i] = p[8 + i]; }
  return r;
}

// B fragment 32x16 (KxN) where B[k][n] = W[n][k] (W row-major NxK): 16
// k-contiguous bf16 per lane. Lanes 0-15: N=lane, K=0..15; 16-31: K=16..31.
static __device__ __forceinline__ bf16x16 load_b_frag_rowk(const __bf16* W, int ldw, int lane) {
  const __bf16* p = W + (size_t)(lane & 15) * ldw + ((lane >> 4) << 4);
  Frag32B u;
  u.q[0] = *(const uint4*)(p);
  u.q[1] = *(const uint4*)(p + 8);
  return u.v;
}

// B fragment 32x16 (KxN) from a row-major KxN source (element-wise; LDS use)
static __device__ __forceinline__ bf16x16 load_b_frag_colk_e(const __bf16* V, int strideK, int lane) {
  int n = lane & 15;
  int kb = (lane >> 4) << 4;
  bf16x16 r;
#pragma unroll
  for (int i = 0; i < 16; ++i) r[i] = V[(size_t)(kb + i) * strideK + n];
  return r;
}

// ---------------------------------------------------------------------------
// K0: fp32 -> bf16 conversion
__global__ __launch_bounds__(256) void f32_to_bf16_k(const float* __restrict__ s,
                                                     __bf16* __restrict__ d, int n) {
  int i = blockIdx.x * 256 + threadIdx.x;
  if (i < n) d[i] = f2bf(s[i]);
}

// ---------------------------------------------------------------------------
// Generic GEMM: D[m,n] = sum_k A[m,k] * W[n,k]  (A bf16 MxK, W bf16 NxK)
// One 16x64 tile per wave (4 accumulators, A-fragment reused 4x).
// Output either bf16 (Dbf) or f32 (Df). Requires N % 64 == 0.
__global__ __launch_bounds__(256) void gemm_xwT_k(const __bf16* __restrict__ A,
                                                  const __bf16* __restrict__ W,
                                                  __bf16* __restrict__ Dbf,
                                                  float* __restrict__ Df,
                                                  int M, int N, int K,
                                                  int lda, int ldw, int ldd) {
  int wave = threadIdx.x >> 5, lane = threadIdx.x & 31;
  int tile = blockIdx.x * 8 + wave;
  int ntn = N >> 6;                 // tiles of 64 columns
  int nt = (tile % ntn) << 2;       // starting 16-col tile index
  int mt = tile / ntn;
  if (mt * 16 >= M) return;
  const __bf16* Ap = A + (size_t)mt * 16 * lda;
  f32x8 acc[4] = {F32X8_ZERO, F32X8_ZERO, F32X8_ZERO, F32X8_ZERO};
  for (int k0 = 0; k0 < K; k0 += 32) {
    if (k0 + 32 < K) __builtin_prefetch(Ap + k0 + 32, 0, 3);  // global_prefetch_b8
    bf16x16 a = load_a_frag_g(Ap + k0, lda, lane);
#pragma unroll
    for (int j = 0; j < 4; ++j) {
      bf16x16 b = load_b_frag_rowk(W + (size_t)(nt + j) * 16 * ldw + k0, ldw, lane);
      acc[j] = wmma_bf16(a, b, acc[j]);
    }
  }
  int n0 = lane & 15;
  int m0 = mt * 16 + ((lane >> 4) << 3);
  if (Df) {
#pragma unroll
    for (int j4 = 0; j4 < 4; ++j4) {
      int n = (nt + j4) * 16 + n0;
#pragma unroll
      for (int j = 0; j < 8; ++j) Df[(size_t)(m0 + j) * ldd + n] = acc[j4][j];
    }
  } else {
#pragma unroll
    for (int j4 = 0; j4 < 4; ++j4) {
      int n = (nt + j4) * 16 + n0;
#pragma unroll
      for (int j = 0; j < 8; ++j) Dbf[(size_t)(m0 + j) * ldd + n] = f2bf(acc[j4][j]);
    }
  }
}

// ---------------------------------------------------------------------------
// K2: logits[b,h,r,c] = NORM * q.k  (single K=32 WMMA per 16x16 tile, D=32),
// scattered into `two` tensor layout (b, c, r, f) with f=h in [0,16)
__global__ __launch_bounds__(256) void logits_two_k(const __bf16* __restrict__ qv,
                                                    const __bf16* __restrict__ kv,
                                                    __bf16* __restrict__ two) {
  int wave = threadIdx.x >> 5, lane = threadIdx.x & 31;
  int t = blockIdx.x * 8 + wave;
  int ct = t & 15; t >>= 4;
  int rt = t & 15; t >>= 4;
  int h  = t & 15; t >>= 4;
  int b  = t;
  const __bf16* Aq = qv + (size_t)(b * RR + rt * 16) * 1024 + h * 32;  // q rows, lda=1024
  const __bf16* Bk = kv + (size_t)(b * CC + ct * 16) * 1024 + h * 32;  // k rows (NxK form)
  f32x8 acc = F32X8_ZERO;
  bf16x16 a  = load_a_frag_g(Aq, 1024, lane);
  bf16x16 bb = load_b_frag_rowk(Bk, 1024, lane);
  acc = wmma_bf16(a, bb, acc);
  int n = lane & 15, mb = (lane >> 4) << 3;
  int c = ct * 16 + n;
#pragma unroll
  for (int q = 0; q < 8; ++q) {
    int r = rt * 16 + mb + q;
    two[((size_t)(b * CC + c) * RR + r) * 32 + h] = f2bf(acc[q] * NORM_F);
  }
}

// K2b: two[b,c,r,16..31] = cost[b,r,c]
__global__ __launch_bounds__(256) void fill_cost_k(const float* __restrict__ cost,
                                                   __bf16* __restrict__ two) {
  size_t i = (size_t)blockIdx.x * 256 + threadIdx.x;  // over (b, c, r)
  if (i >= (size_t)BB * CC * RR) return;
  int r = (int)(i & 255);
  size_t t = i >> 8;
  int c = (int)(t & 255);
  int b = (int)(t >> 8);
  __bf16 cv = f2bf(cost[((size_t)b * RR + r) * CC + c]);
  __bf16* p = two + i * 32 + 16;
#pragma unroll
  for (int j = 0; j < 16; ++j) p[j] = cv;
}

// ---------------------------------------------------------------------------
// K3: fused MLP  mixed = relu(two @ Wms1.T) @ Wms2.T
// Each wave handles 16 rows of `two` (fixed b,c; 16 consecutive r).
// Layer-1 tile (K=32, one WMMA) -> relu -> bf16 via LDS strip -> layer-2
// A-fragment -> accumulate K=512 into two persistent accumulators.
// Writes L1 in (b,h,r,c) and L2 pre-transposed in (b,h,c,r).
__global__ __launch_bounds__(256) void mlp_k(const __bf16* __restrict__ two,
                                             const __bf16* __restrict__ Wms1,
                                             const __bf16* __restrict__ Wms2,
                                             float* __restrict__ L1,
                                             float* __restrict__ L2) {
  __shared__ __bf16 lds[8][16 * 32];
  int wave = threadIdx.x >> 5, lane = threadIdx.x & 31;
  size_t strip = (size_t)blockIdx.x * 8 + wave;
  if (strip >= (size_t)BB * CC * RR / 16) return;
  size_t p0 = strip * 16;                 // p = (b*256 + c)*256 + r
  int r0 = (int)(p0 & 255);
  size_t tt = p0 >> 8;
  int c = (int)(tt & 255);
  int b = (int)(tt >> 8);
  bf16x16 a0 = load_a_frag_g(two + p0 * 32, 32, lane);
  f32x8 acc0 = F32X8_ZERO;
  f32x8 acc1 = F32X8_ZERO;
  __bf16* sl = lds[wave];
  int n = lane & 15, mb = (lane >> 4) << 3;
  for (int jp = 0; jp < 16; ++jp) {      // layer-2 K chunks of 32
#pragma unroll
    for (int t2 = 0; t2 < 2; ++t2) {     // two layer-1 N tiles per chunk
      int j = jp * 2 + t2;
      bf16x16 b1 = load_b_frag_rowk(Wms1 + (size_t)j * 16 * 32, 32, lane);
      f32x8 z = F32X8_ZERO;
      f32x8 hcd = wmma_bf16(a0, b1, z);
#pragma unroll
      for (int q = 0; q < 8; ++q) {
        float v = hcd[q];
        v = v > 0.f ? v : 0.f;           // relu
        sl[(mb + q) * 32 + t2 * 16 + n] = f2bf(v);
      }
    }
    __builtin_amdgcn_wave_barrier();     // keep DS store->load ordering
    bf16x16 a2 = load_a_frag_e(sl, 32, lane);
    __builtin_amdgcn_wave_barrier();
    bf16x16 b2a = load_b_frag_rowk(Wms2 + jp * 32, 512, lane);
    bf16x16 b2b = load_b_frag_rowk(Wms2 + (size_t)16 * 512 + jp * 32, 512, lane);
    acc0 = wmma_bf16(a2, b2a, acc0);     // mixed cols 0..15  -> l1 head h=n
    acc1 = wmma_bf16(a2, b2b, acc1);     // mixed cols 16..31 -> l2 head h=n
  }
#pragma unroll
  for (int q = 0; q < 8; ++q) {
    int r = r0 + mb + q;
    L1[(((size_t)b * HH + n) * RR + r) * CC + c] = acc0[q];
    L2[(((size_t)b * HH + n) * CC + c) * RR + r] = acc1[q];
  }
}

// ---------------------------------------------------------------------------
// K4: row softmax of tanh(l)*10 with mask; one wave per 256-element row.
// transposed=0: row=(b,h,r), mask[b,r,col]; transposed=1: row=(b,h,c), mask[b,col,c]
__global__ __launch_bounds__(256) void softmax_k(const float* __restrict__ L,
                                                 const unsigned char* __restrict__ mask,
                                                 __bf16* __restrict__ P,
                                                 int transposed) {
  int wave = threadIdx.x >> 5, lane = threadIdx.x & 31;
  size_t row = (size_t)blockIdx.x * 8 + wave;
  if (row >= (size_t)BB * HH * 256) return;
  int rr = (int)(row & 255);
  int b = (int)(row >> 12);              // row = ((b*16 + h)*256 + rr)
  const float* Lr = L + row * 256;
  float vals[8];
  float mx = -INFINITY;
#pragma unroll
  for (int i = 0; i < 8; ++i) {
    int j = lane + i * 32;
    float l = tanhf(Lr[j]) * TANH_CLIP_F;
    unsigned char m = transposed ? mask[((size_t)b * 256 + j) * 256 + rr]
                                 : mask[((size_t)b * 256 + rr) * 256 + j];
    if (m) l = -INFINITY;
    vals[i] = l;
    mx = fmaxf(mx, l);
  }
#pragma unroll
  for (int off = 16; off > 0; off >>= 1) mx = fmaxf(mx, __shfl_xor(mx, off, 32));
  float s = 0.f;
#pragma unroll
  for (int i = 0; i < 8; ++i) { float e = __expf(vals[i] - mx); vals[i] = e; s += e; }
#pragma unroll
  for (int off = 16; off > 0; off >>= 1) s += __shfl_xor(s, off, 32);
  float inv = 1.f / s;
  __bf16* Pr = P + row * 256;
#pragma unroll
  for (int i = 0; i < 8; ++i) Pr[lane + i * 32] = f2bf(vals[i] * inv);
}

// ---------------------------------------------------------------------------
// K5: heads = P @ V per (b,h): (256x256)@(256x32). One block per (b,h).
// V tile (256x32 bf16, 16KB) staged to LDS once via gfx1250 async-to-LDS
// (ASYNCcnt + s_wait_asynccnt), then B-fragments come from LDS.
// V element at Vbase[(b*256 + k)*1024 + h*32 + d]. Out comb[b,row,h*32+d].
__global__ __launch_bounds__(256) void combine_k(const __bf16* __restrict__ P,
                                                 const __bf16* __restrict__ Vbase,
                                                 __bf16* __restrict__ comb) {
  __shared__ __attribute__((aligned(16))) __bf16 vlds[256 * 32];   // 16KB
  int tid = threadIdx.x;
  int bh = blockIdx.x;
  int h = bh & 15, b = bh >> 4;

  // Stage V tile: 1024 x 16-byte chunks, 4 per thread.
  {
    const __bf16* src = Vbase + (size_t)b * 256 * 1024 + h * 32;
#pragma unroll
    for (int it = 0; it < 4; ++it) {
      int idx = it * 256 + tid;
      int c = idx >> 2;           // row 0..255
      int q = idx & 3;            // 16B chunk within 64B row
#if USE_ASYNC_LDS
      __builtin_amdgcn_global_load_async_to_lds_b128(
          (as1_i32x4_p)(src + (size_t)c * 1024 + q * 8),
          (as3_i32x4_p)(vlds + c * 32 + q * 8),
          0, 0);
#else
      *(uint4*)(vlds + c * 32 + q * 8) = *(const uint4*)(src + (size_t)c * 1024 + q * 8);
#endif
    }
#if USE_ASYNC_LDS
    __builtin_amdgcn_s_wait_asynccnt(0);
#endif
    __syncthreads();
  }

  int wave = tid >> 5, lane = tid & 31;
  for (int rt = wave; rt < 16; rt += 8) {   // 16 r-tiles over 8 waves
    const __bf16* Ap = P + (((size_t)b * HH + h) * 256 + rt * 16) * 256;
    f32x8 acc0 = F32X8_ZERO;
    f32x8 acc1 = F32X8_ZERO;
    for (int kc = 0; kc < 8; ++kc) {
      bf16x16 a = load_a_frag_g(Ap + kc * 32, 256, lane);
      const __bf16* Vb = vlds + (size_t)kc * 32 * 32;
      bf16x16 b0 = load_b_frag_colk_e(Vb, 32, lane);       // d = 0..15
      bf16x16 b1 = load_b_frag_colk_e(Vb + 16, 32, lane);  // d = 16..31
      acc0 = wmma_bf16(a, b0, acc0);
      acc1 = wmma_bf16(a, b1, acc1);
    }
    int n = lane & 15, mb = (lane >> 4) << 3;
#pragma unroll
    for (int q = 0; q < 8; ++q) {
      size_t r = (size_t)rt * 16 + mb + q;
      __bf16* o = comb + ((size_t)b * 256 + r) * EE + h * 32;
      o[n]      = f2bf(acc0[q]);
      o[16 + n] = f2bf(acc1[q]);
    }
  }
}

// ---------------------------------------------------------------------------
extern "C" void kernel_launch(void* const* d_in, const int* in_sizes, int n_in,
                              void* d_out, int out_size, void* d_ws, size_t ws_size,
                              hipStream_t stream) {
  (void)in_sizes; (void)n_in; (void)out_size; (void)ws_size;
  const float* x1    = (const float*)d_in[0];
  const float* x2    = (const float*)d_in[1];
  const float* cost  = (const float*)d_in[2];
  const unsigned char* mask = (const unsigned char*)d_in[3];
  const float* Wqv1  = (const float*)d_in[4];
  const float* Wkv2  = (const float*)d_in[5];
  const float* Wms1  = (const float*)d_in[6];
  const float* Wms2  = (const float*)d_in[7];
  const float* Wout1 = (const float*)d_in[8];
  const float* Wout2 = (const float*)d_in[9];
  float* out = (float*)d_out;

  char* ws = (char*)d_ws;
  size_t off = 0;
  auto carve = [&](size_t bytes) -> char* {
    char* p = ws + off;
    off += (bytes + 255) & ~(size_t)255;
    return p;
  };

  const size_t nX  = (size_t)BB * RR * EE;      // 524288
  const size_t nQV = (size_t)BB * RR * 1024;    // 1048576
  const size_t nL  = (size_t)BB * HH * RR * CC; // 4194304

  __bf16* x1b    = (__bf16*)carve(nX * 2);
  __bf16* x2b    = (__bf16*)carve(nX * 2);
  __bf16* Wqv1b  = (__bf16*)carve((size_t)1024 * 512 * 2);
  __bf16* Wkv2b  = (__bf16*)carve((size_t)1024 * 512 * 2);
  __bf16* Wms1b  = (__bf16*)carve((size_t)512 * 32 * 2);
  __bf16* Wms2b  = (__bf16*)carve((size_t)32 * 512 * 2);
  __bf16* Wout1b = (__bf16*)carve((size_t)512 * 512 * 2);
  __bf16* Wout2b = (__bf16*)carve((size_t)512 * 512 * 2);
  __bf16* qvb    = (__bf16*)carve(nQV * 2);
  __bf16* kvb    = (__bf16*)carve(nQV * 2);
  __bf16* twob   = (__bf16*)carve((size_t)BB * CC * RR * 32 * 2);
  float*  L1     = (float*)carve(nL * 4);
  float*  L2     = (float*)carve(nL * 4);
  __bf16* P1b    = (__bf16*)carve(nL * 2);
  __bf16* P2b    = (__bf16*)carve(nL * 2);
  __bf16* comb1b = (__bf16*)carve(nX * 2);
  __bf16* comb2b = (__bf16*)carve(nX * 2);

  auto gridN = [](size_t n) { return (unsigned)((n + 255) / 256); };

  // K0: weight/input conversions to bf16
  f32_to_bf16_k<<<gridN(nX), 256, 0, stream>>>(x1, x1b, (int)nX);
  f32_to_bf16_k<<<gridN(nX), 256, 0, stream>>>(x2, x2b, (int)nX);
  f32_to_bf16_k<<<gridN(524288), 256, 0, stream>>>(Wqv1, Wqv1b, 524288);
  f32_to_bf16_k<<<gridN(524288), 256, 0, stream>>>(Wkv2, Wkv2b, 524288);
  f32_to_bf16_k<<<gridN(16384), 256, 0, stream>>>(Wms1, Wms1b, 16384);
  f32_to_bf16_k<<<gridN(16384), 256, 0, stream>>>(Wms2, Wms2b, 16384);
  f32_to_bf16_k<<<gridN(262144), 256, 0, stream>>>(Wout1, Wout1b, 262144);
  f32_to_bf16_k<<<gridN(262144), 256, 0, stream>>>(Wout2, Wout2b, 262144);

  // K1: qv = x1 @ Wqv1.T ; kv = x2 @ Wkv2.T   (M=1024, N=1024, K=512)
  // tiles = (1024/16)*(1024/64) = 1024 -> 128 blocks
  gemm_xwT_k<<<128, 256, 0, stream>>>(x1b, Wqv1b, qvb, nullptr,
                                      1024, 1024, 512, 512, 512, 1024);
  gemm_xwT_k<<<128, 256, 0, stream>>>(x2b, Wkv2b, kvb, nullptr,
                                      1024, 1024, 512, 512, 512, 1024);

  // K2: logits -> two[...,0:16] ; cost -> two[...,16:32]
  logits_two_k<<<2048, 256, 0, stream>>>(qvb, kvb, twob);
  fill_cost_k<<<1024, 256, 0, stream>>>(cost, twob);

  // K3: fused mixed-score MLP (dominant 17 GFLOP)
  mlp_k<<<2048, 256, 0, stream>>>(twob, Wms1b, Wms2b, L1, L2);

  // K4: softmax(tanh*10, mask)
  softmax_k<<<2048, 256, 0, stream>>>(L1, mask, P1b, 0);
  softmax_k<<<2048, 256, 0, stream>>>(L2, mask, P2b, 1);

  // K5: heads1 = P1 @ v2 ; heads2 = P2 @ v1  (v halves live at +512 in qv/kv)
  combine_k<<<64, 256, 0, stream>>>(P1b, kvb + 512, comb1b);
  combine_k<<<64, 256, 0, stream>>>(P2b, qvb + 512, comb2b);

  // K6: h1 = comb1 @ Wout1.T ; h2 = comb2 @ Wout2.T  (f32 outputs)
  // tiles = (1024/16)*(512/64) = 512 -> 64 blocks
  gemm_xwT_k<<<64, 256, 0, stream>>>(comb1b, Wout1b, nullptr, out,
                                     1024, 512, 512, 512, 512, 512);
  gemm_xwT_k<<<64, 256, 0, stream>>>(comb2b, Wout2b, nullptr, out + nX,
                                     1024, 512, 512, 512, 512, 512);
}
